// MoE_21895743275193
// MI455X (gfx1250) — matrix operand order repaired and checked by reading, not compiled, not run on hardware
//
#include <hip/hip_runtime.h>

#define D_MODEL 1024
#define D_FF    4096
#define N_EXP   8
#define M_TILE  32

typedef __attribute__((ext_vector_type(16))) __bf16         v16bf;
typedef __attribute__((ext_vector_type(8)))  float          v8f;
typedef __attribute__((ext_vector_type(4)))  unsigned int   u32x4;
typedef __attribute__((ext_vector_type(8)))  int            i32x8;
typedef __attribute__((ext_vector_type(4)))  int            i32x4;
typedef __attribute__((ext_vector_type(4)))  unsigned short u16x4;

union FragBF16 {            // 32 bytes = 8 VGPRs: one WMMA A or B operand
    v16bf v;
    u32x4 q[2];
};

__device__ __forceinline__ unsigned short f2bf(float f) {
    unsigned u = __builtin_bit_cast(unsigned, f);
    u = u + 0x7FFFu + ((u >> 16) & 1u);        // round-to-nearest-even
    return (unsigned short)(u >> 16);
}

// ---------------------------------------------------------------- convert f32 -> bf16
__global__ __launch_bounds__(256) void moe_cvt_bf16(const float* __restrict__ in,
                                                    unsigned short* __restrict__ out,
                                                    long n4) {
    long i = (long)blockIdx.x * blockDim.x + threadIdx.x;
    long stride = (long)gridDim.x * blockDim.x;
    const float4* in4 = (const float4*)in;
    for (long j = i; j < n4; j += stride) {
        float4 v = in4[j];
        u16x4 o;
        o.x = f2bf(v.x); o.y = f2bf(v.y); o.z = f2bf(v.z); o.w = f2bf(v.w);
        *(u16x4*)(out + j * 4) = o;
    }
}

// ---------------------------------------------------------------- gating: one wave per token
__global__ __launch_bounds__(256) void moe_gate(const float* __restrict__ x,
                                                const float* __restrict__ gate_w,
                                                unsigned short* __restrict__ xbf,
                                                float* __restrict__ imp,
                                                float* __restrict__ probs,
                                                int* __restrict__ cnt,
                                                int* __restrict__ idx_list,
                                                int T) {
    __shared__ float gw[N_EXP * D_MODEL];              // 32 KB
    for (int i = threadIdx.x; i < N_EXP * D_MODEL; i += 256) gw[i] = gate_w[i];
    __syncthreads();

    const int wave = threadIdx.x >> 5, lane = threadIdx.x & 31;
    const int t = blockIdx.x * 8 + wave;
    if (t >= T) return;

    float acc[N_EXP];
#pragma unroll
    for (int e = 0; e < N_EXP; ++e) acc[e] = 0.f;

    const float* xr = x + (size_t)t * D_MODEL;
    for (int d = lane; d < D_MODEL; d += 32) {
        float xv = xr[d];
        xbf[(size_t)t * D_MODEL + d] = f2bf(xv);       // fused x -> bf16
#pragma unroll
        for (int e = 0; e < N_EXP; ++e) acc[e] += xv * gw[e * D_MODEL + d];
    }
#pragma unroll
    for (int e = 0; e < N_EXP; ++e)
#pragma unroll
        for (int off = 16; off > 0; off >>= 1) acc[e] += __shfl_xor(acc[e], off, 32);

    if (lane == 0) {
        float m = acc[0]; int sel = 0;
#pragma unroll
        for (int e = 1; e < N_EXP; ++e) if (acc[e] > m) { m = acc[e]; sel = e; }
        float p[N_EXP], sum = 0.f;
#pragma unroll
        for (int e = 0; e < N_EXP; ++e) { p[e] = __expf(acc[e] - m); sum += p[e]; }
        float inv = 1.f / sum;
#pragma unroll
        for (int e = 0; e < N_EXP; ++e) probs[(size_t)t * N_EXP + e] = p[e] * inv;
        imp[t] = inv;                                  // top-1 prob = exp(0)/sum
        int pos = atomicAdd(&cnt[sel], 1);
        idx_list[sel * T + pos] = t;
    }
}

// ---------------------------------------------------------------- deterministic prob-fraction sum
__global__ __launch_bounds__(256) void moe_probsum(const float* __restrict__ probs,
                                                   float* __restrict__ prob_sum, int T) {
    __shared__ float red[256];
    const int e = blockIdx.x;
    float a = 0.f;
    for (int t = threadIdx.x; t < T; t += 256) a += probs[(size_t)t * N_EXP + e];
    red[threadIdx.x] = a; __syncthreads();
    for (int s = 128; s > 0; s >>= 1) {
        if (threadIdx.x < s) red[threadIdx.x] += red[threadIdx.x + s];
        __syncthreads();
    }
    if (threadIdx.x == 0) prob_sum[e] = red[0];
}

__global__ void moe_loss(const int* __restrict__ cnt, const float* __restrict__ ps,
                         float* __restrict__ loss_out, int T) {
    float acc = 0.f;
#pragma unroll
    for (int e = 0; e < N_EXP; ++e) acc += (float)cnt[e] * ps[e];
    loss_out[0] = acc * (float)N_EXP / ((float)T * (float)T);
}

// ---------------------------------------------------------------- fused expert FFN tile kernel
// grid: (tiles, experts); block: 256 threads = 8 waves; 32 tokens per block.
__global__ __launch_bounds__(256) void moe_ffn(const unsigned short* __restrict__ xbf,
                                               const unsigned short* __restrict__ fc1bf,
                                               const unsigned short* __restrict__ fc2bf,
                                               const float* __restrict__ fc1_b,
                                               const float* __restrict__ fc2_b,
                                               const float* __restrict__ imp,
                                               const int* __restrict__ cnt,
                                               const int* __restrict__ idx_list,
                                               float* __restrict__ out,
                                               int T) {
    const int e    = blockIdx.y;
    const int tile = blockIdx.x;
    const int cn   = cnt[e];
    if (tile * M_TILE >= cn) return;

    __shared__ unsigned short xA[M_TILE * D_MODEL];    // 64 KB: bf16 A tile (32 tokens x 1024)
    __shared__ unsigned short hB[M_TILE * 128];        //  8 KB: bf16 h chunk (32 tokens x 128)
    __shared__ int   tk[M_TILE];
    __shared__ float timp[M_TILE];

    const int tid  = threadIdx.x;
    const int wave = tid >> 5, lane = tid & 31;
    const int n16  = lane & 15;                        // N / M-column index inside fragment
    const int hi   = lane >> 4;                        // lane half select
    const int kbA  = hi * 8;                           // A fragment K base (ISA 7.12.2)
    const int kbB  = hi * 8;                           // B fragment K base

    if (tid < M_TILE) {
        int idx = tile * M_TILE + tid;
        int tok = idx_list[e * T + ((idx < cn) ? idx : tile * M_TILE)]; // pad with first token
        tk[tid] = tok;
        timp[tid] = imp[tok];
    }
    __syncthreads();

    // ---- TDM gather: two descriptors, 16 token rows (1024 bf16 each) per descriptor ----
    if (tid < 32) {                                    // wave 0 only (TDM ignores EXEC)
#pragma unroll
        for (int rb = 0; rb < 2; ++rb) {
            unsigned long long ga = (unsigned long long)(size_t)xbf;
            u32x4 g0;
            g0.x = 1u | (1u << 31);                    // count=1, gather_mode=1, 16-bit idx
            g0.y = (unsigned)(size_t)&xA[rb * 16 * D_MODEL];        // lds_addr
            g0.z = (unsigned)(ga & 0xFFFFFFFFull);                  // global_addr[31:0]
            g0.w = (unsigned)((ga >> 32) & 0x1FFFFFFull) | (2u << 30); // addr[56:32] | type=2
            i32x8 g1;
            g1[0] = (1 << 16);                         // wg_mask=0, data_size=2B
            g1[1] = (int)((unsigned)D_MODEL << 16);    // tensor_dim0[15:0] @ [63:48]
            g1[2] = (int)((unsigned)(T & 0xFFFF) << 16); // tensor_dim1[15:0] @ [95:80]
            g1[3] = (int)((unsigned)D_MODEL << 16) | ((T >> 16) & 0xFFFF); // tile_dim0 | dim1 hi
            g1[4] = 16;                                // tile_dim1 = #row indices
            g1[5] = D_MODEL;                           // tensor_dim0_stride (elements)
            g1[6] = 0;
            g1[7] = 0;
            i32x4 g2, g3;
#pragma unroll
            for (int i = 0; i < 4; ++i) {
                g2[i] = (tk[rb * 16 + 2 * i] & 0xFFFF) | (tk[rb * 16 + 2 * i + 1] << 16);
                g3[i] = (tk[rb * 16 + 8 + 2 * i] & 0xFFFF) | (tk[rb * 16 + 9 + 2 * i] << 16);
            }
            i32x8 g4 = (i32x8){0, 0, 0, 0, 0, 0, 0, 0};   // unused trailing group
            __builtin_amdgcn_tensor_load_to_lds(g0, g1, g2, g3, g4, 0);
        }
        __builtin_amdgcn_s_wait_tensorcnt(0);
    }
    __syncthreads();

    // ---- persistent y accumulators: wave w owns out columns [w*128, w*128+128) ----
    v8f cy[2][8];
#pragma unroll
    for (int rb = 0; rb < 2; ++rb)
#pragma unroll
        for (int j = 0; j < 8; ++j) cy[rb][j] = (v8f){0.f,0.f,0.f,0.f,0.f,0.f,0.f,0.f};

    for (int fc = 0; fc < D_FF; fc += 128) {
        // ===== GEMM1: h(32x16) = x(32x1024) * fc1^T, this wave's f block =====
        const int f0 = fc + wave * 16;
        v8f ch0 = (v8f){0.f,0.f,0.f,0.f,0.f,0.f,0.f,0.f};
        v8f ch1 = ch0;
        const unsigned short* w1 = fc1bf + ((size_t)e * D_FF + f0 + n16) * D_MODEL;
#pragma unroll 4
        for (int kk = 0; kk < D_MODEL; kk += 32) {
            FragBF16 a0, a1, b;
            b.q[0]  = *(const u32x4*)&w1[kk + kbB];
            b.q[1]  = *(const u32x4*)&w1[kk + kbB + 16];
            a0.q[0] = *(const u32x4*)&xA[n16 * D_MODEL + kk + kbA];
            a0.q[1] = *(const u32x4*)&xA[n16 * D_MODEL + kk + kbA + 16];
            a1.q[0] = *(const u32x4*)&xA[(16 + n16) * D_MODEL + kk + kbA];
            a1.q[1] = *(const u32x4*)&xA[(16 + n16) * D_MODEL + kk + kbA + 16];
            ch0 = __builtin_amdgcn_wmma_f32_16x16x32_bf16(false, a0.v, false, b.v,
                                                          (short)0, ch0, false, false);
            ch1 = __builtin_amdgcn_wmma_f32_16x16x32_bf16(false, a1.v, false, b.v,
                                                          (short)0, ch1, false, false);
        }
        // relu + bias, write bf16 h chunk to LDS
        const float b1 = fc1_b[e * D_FF + f0 + n16];
#pragma unroll
        for (int r = 0; r < 8; ++r) {
            int m = r + 8 * hi;
            hB[m * 128 + wave * 16 + n16]        = f2bf(fmaxf(ch0[r] + b1, 0.f));
            hB[(16 + m) * 128 + wave * 16 + n16] = f2bf(fmaxf(ch1[r] + b1, 0.f));
        }
        __syncthreads();

        // ===== GEMM2: y(32x128 cols of this wave) += h(32x128) * fc2^T =====
#pragma unroll
        for (int j = 0; j < 8; ++j) {
            const int dc0 = wave * 128 + j * 16;
            const unsigned short* w2 = fc2bf + ((size_t)e * D_MODEL + dc0 + n16) * D_FF + fc;
#pragma unroll
            for (int kk = 0; kk < 128; kk += 32) {
                FragBF16 a0, a1, b;
                b.q[0]  = *(const u32x4*)&w2[kk + kbB];
                b.q[1]  = *(const u32x4*)&w2[kk + kbB + 16];
                a0.q[0] = *(const u32x4*)&hB[n16 * 128 + kk + kbA];
                a0.q[1] = *(const u32x4*)&hB[n16 * 128 + kk + kbA + 16];
                a1.q[0] = *(const u32x4*)&hB[(16 + n16) * 128 + kk + kbA];
                a1.q[1] = *(const u32x4*)&hB[(16 + n16) * 128 + kk + kbA + 16];
                cy[0][j] = __builtin_amdgcn_wmma_f32_16x16x32_bf16(false, a0.v, false, b.v,
                                                                   (short)0, cy[0][j], false, false);
                cy[1][j] = __builtin_amdgcn_wmma_f32_16x16x32_bf16(false, a1.v, false, b.v,
                                                                   (short)0, cy[1][j], false, false);
            }
        }
        __syncthreads();                               // hB reused next chunk
    }

    // ---- epilogue: bias, gate scale, scatter (each token routed exactly once) ----
    const int nvalid = cn - tile * M_TILE;
#pragma unroll
    for (int j = 0; j < 8; ++j) {
        const int dcol = wave * 128 + j * 16 + n16;
        const float b2 = fc2_b[e * D_MODEL + dcol];
#pragma unroll
        for (int rb = 0; rb < 2; ++rb)
#pragma unroll
            for (int r = 0; r < 8; ++r) {
                int m = rb * 16 + r + 8 * hi;
                if (m < nvalid)
                    out[(size_t)tk[m] * D_MODEL + dcol] = (cy[rb][j][r] + b2) * timp[m];
            }
    }
}

// ---------------------------------------------------------------- host launch
static inline size_t alignup(size_t v) { return (v + 255) & ~(size_t)255; }

extern "C" void kernel_launch(void* const* d_in, const int* in_sizes, int n_in,
                              void* d_out, int out_size, void* d_ws, size_t ws_size,
                              hipStream_t stream) {
    const float* x      = (const float*)d_in[0];
    const float* gate_w = (const float*)d_in[1];
    const float* fc1_w  = (const float*)d_in[2];
    const float* fc1_b  = (const float*)d_in[3];
    const float* fc2_w  = (const float*)d_in[4];
    const float* fc2_b  = (const float*)d_in[5];
    float* out = (float*)d_out;

    const int T = in_sizes[0] / D_MODEL;

    // workspace carve-up (256B aligned)
    char* ws = (char*)d_ws;
    size_t o = 0;
    unsigned short* xbf   = (unsigned short*)(ws + o); o = alignup(o + (size_t)T * D_MODEL * 2);
    unsigned short* fc1bf = (unsigned short*)(ws + o); o = alignup(o + (size_t)N_EXP * D_FF * D_MODEL * 2);
    unsigned short* fc2bf = (unsigned short*)(ws + o); o = alignup(o + (size_t)N_EXP * D_MODEL * D_FF * 2);
    float* imp            = (float*)(ws + o);          o = alignup(o + (size_t)T * 4);
    int*   idx_list       = (int*)(ws + o);            o = alignup(o + (size_t)N_EXP * T * 4);
    int*   cnt            = (int*)(ws + o);            o = alignup(o + 256);
    float* prob_sum       = (float*)(ws + o);          o = alignup(o + 256);
    float* probs          = (float*)(ws + o);          o = alignup(o + (size_t)T * N_EXP * 4);

    (void)hipMemsetAsync(cnt, 0, 256, stream);

    // weights -> bf16 (L2-resident working set afterwards)
    const long nW4 = (long)N_EXP * D_FF * D_MODEL / 4;
    moe_cvt_bf16<<<4096, 256, 0, stream>>>(fc1_w, fc1bf, nW4);
    moe_cvt_bf16<<<4096, 256, 0, stream>>>(fc2_w, fc2bf, nW4);

    // gating + routing lists + fused x->bf16
    moe_gate<<<(T + 7) / 8, 256, 0, stream>>>(x, gate_w, xbf, imp, probs, cnt, idx_list, T);

    // deterministic load-balancing loss
    moe_probsum<<<N_EXP, 256, 0, stream>>>(probs, prob_sum, T);
    moe_loss<<<1, 1, 0, stream>>>(cnt, prob_sum, out + (size_t)T * D_MODEL, T);

    // fused expert FFN
    dim3 grid((T + M_TILE - 1) / M_TILE, N_EXP);
    moe_ffn<<<grid, 256, 0, stream>>>(xbf, fc1bf, fc2bf, fc1_b, fc2_b, imp,
                                      cnt, idx_list, out, T);
}
